// SpatialFusionLayer_22119081574818
// MI455X (gfx1250) — matrix operand order repaired and checked by reading, not compiled, not run on hardware
//
#include <hip/hip_runtime.h>

#define B_   2
#define N_   19560
#define BN_  39120      // B_*N_, 16*2445
#define C_   256
#define M_   8
#define L_   4
#define P_   4
#define DH_  32
#define FF_  1024
#define KC_  64         // TDM K-chunk (elements)

typedef __attribute__((ext_vector_type(16))) __bf16 v16bf;
typedef __attribute__((ext_vector_type(8)))  __bf16 v8bf;
typedef __attribute__((ext_vector_type(8)))  float  v8f;

#if defined(__HIP_DEVICE_COMPILE__) && defined(__gfx1250__)
# if __has_builtin(__builtin_amdgcn_tensor_load_to_lds)
#  define USE_TDM 1
# endif
#endif
#ifndef USE_TDM
# define USE_TDM 0
#endif

#if USE_TDM
typedef unsigned int u32x4 __attribute__((ext_vector_type(4)));
typedef int          i32x4 __attribute__((ext_vector_type(4)));
typedef int          i32x8 __attribute__((ext_vector_type(8)));

// DMA a 64-row x KC_-col (elements) slab of Bt (row stride = Kdim elements,
// 2-byte data) into LDS at byte offset lds_off. D# per CDNA5 ISA ch.8.
__device__ __forceinline__ void tdm_load_b(const __bf16* gsrc, unsigned lds_off, int Kdim) {
    unsigned long long ga = (unsigned long long)(uintptr_t)gsrc;
    u32x4 g0 = { 1u,                                   // count=1 valid descriptor
                 lds_off,                               // lds_addr
                 (unsigned)(ga & 0xffffffffu),          // global_addr[31:0]
                 (unsigned)((ga >> 32) & 0x1ffffffu) | (2u << 30) };  // addr[56:32] | type=2
    i32x8 g1 = { (int)(1u << 16),                      // data_size=1 (2 bytes)
                 (int)((Kdim & 0xffff) << 16),         // tensor_dim0[15:0]
                 (int)((((unsigned)Kdim >> 16) & 0xffffu) | (64u << 16)), // dim0[31:16]|tensor_dim1=64
                 (int)(KC_ << 16),                     // tile_dim0 = KC_
                 64,                                   // tile_dim1 = 64
                 Kdim,                                 // tensor_dim0_stride[31:0]
                 0, 0 };
    i32x4 gz = { 0, 0, 0, 0 };
# if __has_include(<hip/amd_detail/amd_gfx1250_TDM.h>)
    i32x8 gz8 = { 0, 0, 0, 0, 0, 0, 0, 0 };
    __builtin_amdgcn_tensor_load_to_lds(g0, g1, gz, gz, gz8, 0);
# else
    __builtin_amdgcn_tensor_load_to_lds(g0, g1, gz, gz, 0);
# endif
}
#endif

__device__ __forceinline__ __bf16 f2bf(float f) {
    unsigned u = __builtin_bit_cast(unsigned, f);
    unsigned r = u + 0x7FFFu + ((u >> 16) & 1u);   // round-to-nearest-even
    unsigned short h = (unsigned short)(r >> 16);
    return __builtin_bit_cast(__bf16, h);
}
__device__ __forceinline__ float bf2f(__bf16 b) {
    unsigned short h = __builtin_bit_cast(unsigned short, b);
    unsigned u = ((unsigned)h) << 16;
    return __builtin_bit_cast(float, u);
}

// ---------------- prep: q = bf16(src+pos), s = bf16(src) ----------------
__global__ void k_prep(const float* __restrict__ src, const float* __restrict__ pos,
                       __bf16* __restrict__ sbf, __bf16* __restrict__ qbf, long n) {
    long i = (long)blockIdx.x * blockDim.x + threadIdx.x;
    if (i < n) {
        float s = src[i];
        sbf[i] = f2bf(s);
        qbf[i] = f2bf(s + pos[i]);
    }
}

// ------- weight transpose+convert: W[K][Nw] f32 -> Wt[Nw][K] bf16 -------
__global__ void k_wt(const float* __restrict__ W, __bf16* __restrict__ Wt, int K, int Nw) {
    long i = (long)blockIdx.x * blockDim.x + threadIdx.x;
    if (i < (long)K * Nw) {
        int k = (int)(i / Nw), c = (int)(i % Nw);
        Wt[(size_t)c * K + k] = f2bf(W[i]);
    }
}

// ---------------- WMMA GEMM: one wave, 16 rows x 64 cols ----------------
// A: [rows][Kdim] bf16 row-major.  Bt: [Nw][Kdim] bf16 (B column-major).
// B panels staged LDS-side via Tensor Data Mover when available (double
// buffered, pipelined on TENSORcnt); A streamed from global/L2.
enum { EPI_F32 = 0, EPI_F32_RESID = 1, EPI_BF16 = 2, EPI_BF16_RELU = 3 };

template <int EPI>
__global__ void k_gemm_wmma(const __bf16* __restrict__ A, const __bf16* __restrict__ Bt,
                            const float* __restrict__ bias, const float* __restrict__ resid,
                            float* __restrict__ Cf, __bf16* __restrict__ Cb,
                            int Kdim, int Nw) {
    const int lane = threadIdx.x;          // blockDim.x == 32, full wave, EXEC all ones
    const int r    = lane & 15;
    const int h    = lane >> 4;
    const long colBase = (long)blockIdx.x * 64;
    const __bf16* arow = A + ((long)blockIdx.y * 16 + r) * (long)Kdim;

    v8f acc[4] = {};

#if USE_TDM
    __shared__ __attribute__((aligned(32))) __bf16 lds_b[2][64 * KC_];  // 2 x 8KB
    const unsigned lds0 = (unsigned)(uintptr_t)&lds_b[0][0];
    const unsigned lds1 = (unsigned)(uintptr_t)&lds_b[1][0];
    const __bf16* bbase = Bt + colBase * (long)Kdim;

    tdm_load_b(bbase, lds0, Kdim);
    int buf = 0;
    for (int k0 = 0; k0 < Kdim; k0 += KC_) {
        if (k0 + KC_ < Kdim) {                       // prefetch next chunk into other buffer
            tdm_load_b(bbase + (k0 + KC_), buf ? lds0 : lds1, Kdim);
            __builtin_amdgcn_s_wait_tensorcnt(1);    // current chunk resident
        } else {
            __builtin_amdgcn_s_wait_tensorcnt(0);
        }
        __builtin_prefetch(arow + k0 + KC_, 0, 3);
        const __bf16* bp = &lds_b[buf][0];           // LDS layout: [col][k], k contiguous
#pragma unroll
        for (int kk = 0; kk < KC_; kk += 32) {
            v8bf a0 = *(const v8bf*)(arow + k0 + kk + h * 16);
            v8bf a1 = *(const v8bf*)(arow + k0 + kk + h * 16 + 8);
            v16bf a = __builtin_shufflevector(a0, a1, 0,1,2,3,4,5,6,7,8,9,10,11,12,13,14,15);
#pragma unroll
            for (int c = 0; c < 4; ++c) {
                v16bf b = *(const v16bf*)(bp + (c * 16 + r) * KC_ + kk + h * 16);
                acc[c] = __builtin_amdgcn_wmma_f32_16x16x32_bf16(false, a, false, b,
                                                                 (short)0, acc[c], false, false);
            }
        }
        buf ^= 1;
    }
#else
    const __bf16* brow = Bt + (colBase + r) * (long)Kdim;   // + c*16*Kdim per column group
    for (int k0 = 0; k0 < Kdim; k0 += 32) {
        v8bf a0 = *(const v8bf*)(arow + k0 + h * 16);
        v8bf a1 = *(const v8bf*)(arow + k0 + h * 16 + 8);
        v16bf a = __builtin_shufflevector(a0, a1, 0,1,2,3,4,5,6,7,8,9,10,11,12,13,14,15);
        v16bf b0 = *(const v16bf*)(brow + (long)0 * 16 * Kdim + k0 + h * 16);
        v16bf b1 = *(const v16bf*)(brow + (long)1 * 16 * Kdim + k0 + h * 16);
        v16bf b2 = *(const v16bf*)(brow + (long)2 * 16 * Kdim + k0 + h * 16);
        v16bf b3 = *(const v16bf*)(brow + (long)3 * 16 * Kdim + k0 + h * 16);
        if (k0 + 32 < Kdim) __builtin_prefetch(arow + k0 + 32, 0, 3);
        acc[0] = __builtin_amdgcn_wmma_f32_16x16x32_bf16(false, a, false, b0, (short)0, acc[0], false, false);
        acc[1] = __builtin_amdgcn_wmma_f32_16x16x32_bf16(false, a, false, b1, (short)0, acc[1], false, false);
        acc[2] = __builtin_amdgcn_wmma_f32_16x16x32_bf16(false, a, false, b2, (short)0, acc[2], false, false);
        acc[3] = __builtin_amdgcn_wmma_f32_16x16x32_bf16(false, a, false, b3, (short)0, acc[3], false, false);
    }
#endif

#pragma unroll
    for (int c = 0; c < 4; ++c) {
        const int   ccol = (int)colBase + c * 16 + r;
        const float bv   = bias[ccol];
#pragma unroll
        for (int i = 0; i < 8; ++i) {
            const long  crow = (long)blockIdx.y * 16 + i + 8 * h;  // D: row = vgpr + 8*half
            const size_t idx = (size_t)crow * Nw + ccol;
            float v = acc[c][i] + bv;
            if (EPI == EPI_F32)            Cf[idx] = v;
            else if (EPI == EPI_F32_RESID) Cf[idx] = v + resid[idx];
            else if (EPI == EPI_BF16)      Cb[idx] = f2bf(v);
            else                           Cb[idx] = f2bf(fmaxf(v, 0.0f));
        }
    }
}

// ---------------- softmax over 16 (L*P) per (token, head) ----------------
__global__ void k_softmax(float* __restrict__ attn, long nrows) {
    long rr = (long)blockIdx.x * blockDim.x + threadIdx.x;
    if (rr >= nrows) return;
    float* p = attn + rr * 16;
    float mx = p[0];
#pragma unroll
    for (int i = 1; i < 16; ++i) mx = fmaxf(mx, p[i]);
    float s = 0.f;
    float e[16];
#pragma unroll
    for (int i = 0; i < 16; ++i) { e[i] = __expf(p[i] - mx); s += e[i]; }
    float inv = 1.0f / s;
#pragma unroll
    for (int i = 0; i < 16; ++i) p[i] = e[i] * inv;
}

// ---------------- deformable bilinear sampling ---------------------------
// grid = BN_ blocks, blockDim = 256: thread = (m, dh). Output [token][m*32+dh] bf16.
__global__ void k_sample(const __bf16* __restrict__ val, const float* __restrict__ off,
                         const float* __restrict__ attn, const float* __restrict__ refp,
                         const int* __restrict__ shapes, const int* __restrict__ lsi,
                         __bf16* __restrict__ out) {
    const int token = blockIdx.x;
    const int b  = token / N_;
    const int m  = threadIdx.x >> 5;
    const int dh = threadIdx.x & 31;
    const float* offp = off  + (size_t)token * 256 + m * 32;   // [l*8 + p*2 + c]
    const float* attp = attn + (size_t)token * 128 + m * 16;   // [l*4 + p]
    const __bf16* vbase = val + ((size_t)b * N_) * C_ + m * DH_ + dh;

    float acc = 0.f;
    for (int l = 0; l < L_; ++l) {
        const int H  = shapes[l * 2 + 0];
        const int W  = shapes[l * 2 + 1];
        const int s0 = lsi[l];
        const float rx = refp[((size_t)token * L_ + l) * 2 + 0];
        const float ry = refp[((size_t)token * L_ + l) * 2 + 1];
        for (int p = 0; p < P_; ++p) {
            const float locx = rx + offp[l * 8 + p * 2 + 0] / (float)W;
            const float locy = ry + offp[l * 8 + p * 2 + 1] / (float)H;
            const float px = locx * (float)W - 0.5f;
            const float py = locy * (float)H - 0.5f;
            const float x0f = floorf(px), y0f = floorf(py);
            const float wx = px - x0f, wy = py - y0f;
            const int x0 = (int)x0f, y0 = (int)y0f;
            float c00 = 0.f, c10 = 0.f, c01 = 0.f, c11 = 0.f;
            const bool vx0 = (x0 >= 0) & (x0 < W);
            const bool vx1 = (x0 + 1 >= 0) & (x0 + 1 < W);
            const bool vy0 = (y0 >= 0) & (y0 < H);
            const bool vy1 = (y0 + 1 >= 0) & (y0 + 1 < H);
            if (vy0) {
                const size_t rowb = (size_t)(s0 + y0 * W);
                if (vx0) c00 = bf2f(vbase[(rowb + x0) * C_]);
                if (vx1) c10 = bf2f(vbase[(rowb + x0 + 1) * C_]);
            }
            if (vy1) {
                const size_t rowb = (size_t)(s0 + (y0 + 1) * W);
                if (vx0) c01 = bf2f(vbase[(rowb + x0) * C_]);
                if (vx1) c11 = bf2f(vbase[(rowb + x0 + 1) * C_]);
            }
            const float samp = c00 * (1.f - wx) * (1.f - wy) + c10 * wx * (1.f - wy)
                             + c01 * (1.f - wx) * wy         + c11 * wx * wy;
            acc += attp[l * 4 + p] * samp;
        }
    }
    out[(size_t)token * C_ + m * DH_ + dh] = f2bf(acc);
}

// ---------------- layernorm kernels (one token per 256-thread block) -----
__device__ __forceinline__ void block_meanvar(float v, float& mean, float& var,
                                              float* smem /* [16] */) {
    float s1 = v, s2 = v * v;
#pragma unroll
    for (int o = 16; o > 0; o >>= 1) { s1 += __shfl_xor(s1, o, 32); s2 += __shfl_xor(s2, o, 32); }
    const int w = threadIdx.x >> 5;
    if ((threadIdx.x & 31) == 0) { smem[w] = s1; smem[8 + w] = s2; }
    __syncthreads();
    float m = 0.f, q = 0.f;
#pragma unroll
    for (int i = 0; i < 8; ++i) { m += smem[i]; q += smem[8 + i]; }
    m *= (1.0f / C_);
    q = q * (1.0f / C_) - m * m;
    mean = m; var = q;
}

__global__ void k_ln1(float* __restrict__ x32, __bf16* __restrict__ xbf,
                      const float* __restrict__ g, const float* __restrict__ bta) {
    __shared__ float smem[16];
    const size_t idx = (size_t)blockIdx.x * C_ + threadIdx.x;
    const float v = x32[idx];
    float m, q;
    block_meanvar(v, m, q, smem);
    const float y = (v - m) * rsqrtf(q + 1e-5f) * g[threadIdx.x] + bta[threadIdx.x];
    x32[idx] = y;
    xbf[idx] = f2bf(y);
}

__global__ void k_ln2(const float* __restrict__ x32, const float* __restrict__ ffn,
                      const float* __restrict__ g, const float* __restrict__ bta,
                      float* __restrict__ out) {
    __shared__ float smem[16];
    const size_t idx = (size_t)blockIdx.x * C_ + threadIdx.x;
    const float v = x32[idx] + ffn[idx];
    float m, q;
    block_meanvar(v, m, q, smem);
    out[idx] = (v - m) * rsqrtf(q + 1e-5f) * g[threadIdx.x] + bta[threadIdx.x];
}

// ------------------------------- launch ----------------------------------
extern "C" void kernel_launch(void* const* d_in, const int* in_sizes, int n_in,
                              void* d_out, int out_size, void* d_ws, size_t ws_size,
                              hipStream_t stream) {
    const float* src    = (const float*)d_in[0];
    const float* pos    = (const float*)d_in[1];
    const float* refp   = (const float*)d_in[2];
    const int*   shapes = (const int*)d_in[3];
    const int*   lsi    = (const int*)d_in[4];
    const float* W_off  = (const float*)d_in[5];
    const float* b_off  = (const float*)d_in[6];
    const float* W_attn = (const float*)d_in[7];
    const float* b_attn = (const float*)d_in[8];
    const float* W_val  = (const float*)d_in[9];
    const float* b_val  = (const float*)d_in[10];
    const float* W_out  = (const float*)d_in[11];
    const float* b_out  = (const float*)d_in[12];
    const float* ln1_g  = (const float*)d_in[13];
    const float* ln1_b  = (const float*)d_in[14];
    const float* W1     = (const float*)d_in[15];
    const float* b1     = (const float*)d_in[16];
    const float* W2     = (const float*)d_in[17];
    const float* b2     = (const float*)d_in[18];
    const float* ln2_g  = (const float*)d_in[19];
    const float* ln2_b  = (const float*)d_in[20];
    float* out = (float*)d_out;

    // workspace carve-up (256B aligned)
    size_t woff = 0;
    auto give = [&](size_t bytes) -> void* {
        void* p = (char*)d_ws + woff;
        woff += (bytes + 255) & ~(size_t)255;
        return p;
    };
    __bf16* s_bf   = (__bf16*)give((size_t)BN_ * C_ * 2);   // src bf16, later x bf16
    __bf16* q_bf   = (__bf16*)give((size_t)BN_ * C_ * 2);   // query bf16, later acc bf16
    __bf16* val_bf = (__bf16*)give((size_t)BN_ * C_ * 2);
    float*  off_f  = (float*) give((size_t)BN_ * 256 * 4);  // offsets, later ffn out
    float*  attn_f = (float*) give((size_t)BN_ * 128 * 4);
    float*  x32    = (float*) give((size_t)BN_ * C_ * 4);   // src2 / x
    __bf16* h_bf   = (__bf16*)give((size_t)BN_ * FF_ * 2);
    __bf16* Wv_t   = (__bf16*)give((size_t)C_ * C_ * 2);
    __bf16* Woff_t = (__bf16*)give((size_t)C_ * C_ * 2);
    __bf16* Wa_t   = (__bf16*)give((size_t)128 * C_ * 2);
    __bf16* Wout_t = (__bf16*)give((size_t)C_ * C_ * 2);
    __bf16* W1_t   = (__bf16*)give((size_t)FF_ * C_ * 2);
    __bf16* W2_t   = (__bf16*)give((size_t)C_ * FF_ * 2);

    const long nAct = (long)BN_ * C_;
    k_prep<<<dim3((unsigned)(nAct / 256)), dim3(256), 0, stream>>>(src, pos, s_bf, q_bf, nAct);

    k_wt<<<dim3(C_ * C_ / 256),  dim3(256), 0, stream>>>(W_val,  Wv_t,   C_,  C_);
    k_wt<<<dim3(C_ * C_ / 256),  dim3(256), 0, stream>>>(W_off,  Woff_t, C_,  C_);
    k_wt<<<dim3(C_ * 128 / 256), dim3(256), 0, stream>>>(W_attn, Wa_t,   C_,  128);
    k_wt<<<dim3(C_ * C_ / 256),  dim3(256), 0, stream>>>(W_out,  Wout_t, C_,  C_);
    k_wt<<<dim3(C_ * FF_ / 256), dim3(256), 0, stream>>>(W1,     W1_t,   C_,  FF_);
    k_wt<<<dim3(FF_ * C_ / 256), dim3(256), 0, stream>>>(W2,     W2_t,   FF_, C_);

    const unsigned mt = BN_ / 16;  // 2445 row tiles
    // value = src @ W_val + b_val            -> bf16 [token][m*32+dh]
    k_gemm_wmma<EPI_BF16><<<dim3(C_ / 64, mt), dim3(32), 0, stream>>>(
        s_bf, Wv_t, b_val, nullptr, nullptr, val_bf, C_, C_);
    // off = query @ W_off + b_off            -> f32
    k_gemm_wmma<EPI_F32><<<dim3(C_ / 64, mt), dim3(32), 0, stream>>>(
        q_bf, Woff_t, b_off, nullptr, off_f, nullptr, C_, C_);
    // attn logits = query @ W_attn + b_attn  -> f32
    k_gemm_wmma<EPI_F32><<<dim3(128 / 64, mt), dim3(32), 0, stream>>>(
        q_bf, Wa_t, b_attn, nullptr, attn_f, nullptr, C_, 128);

    k_softmax<<<dim3((BN_ * M_ + 255) / 256), dim3(256), 0, stream>>>(attn_f, (long)BN_ * M_);

    // sampling -> acc (reuse q_bf)
    k_sample<<<dim3(BN_), dim3(256), 0, stream>>>(val_bf, off_f, attn_f, refp, shapes, lsi, q_bf);

    // src2 = acc @ W_out + b_out + src -> x32
    k_gemm_wmma<EPI_F32_RESID><<<dim3(C_ / 64, mt), dim3(32), 0, stream>>>(
        q_bf, Wout_t, b_out, src, x32, nullptr, C_, C_);

    // x = LN1(src2) -> x32 (f32) and s_bf (bf16, reuse)
    k_ln1<<<dim3(BN_), dim3(256), 0, stream>>>(x32, s_bf, ln1_g, ln1_b);

    // h = relu(x @ W1 + b1) -> bf16
    k_gemm_wmma<EPI_BF16_RELU><<<dim3(FF_ / 64, mt), dim3(32), 0, stream>>>(
        s_bf, W1_t, b1, nullptr, nullptr, h_bf, C_, FF_);
    // ffn = h @ W2 + b2 -> f32 (reuse off_f)
    k_gemm_wmma<EPI_F32><<<dim3(C_ / 64, mt), dim3(32), 0, stream>>>(
        h_bf, W2_t, b2, nullptr, off_f, nullptr, FF_, C_);

    // out = LN2(x + ffn)
    k_ln2<<<dim3(BN_), dim3(256), 0, stream>>>(x32, off_f, ln2_g, ln2_b, out);

    (void)in_sizes; (void)n_in; (void)out_size; (void)ws_size;
}
